// MDCN3x3Norm_62740882260588
// MI455X (gfx1250) — matrix operand order repaired and checked by reading, not compiled, not run on hardware
//
#include <hip/hip_runtime.h>

// ---------------------------------------------------------------------------
// MDCN 3x3 (DCNv2) + GroupNorm, fused deformable-im2col -> bf16x3 WMMA GEMM.
// Target: gfx1250 (MI455X), wave32, v_wmma_f32_16x16x32_bf16.
// Round 2: double-buffered V chunk + software-pipelined gather (loads for
// chunk s+1 in flight during WMMAs of chunk s), one barrier per k-step.
// ---------------------------------------------------------------------------

typedef __attribute__((ext_vector_type(16))) __bf16 v16bf;
typedef __attribute__((ext_vector_type(8)))  float  v8f;

#define Hc   96
#define Wc   96
#define HWc  9216
#define Cc   256
#define Oc   256
#define CKc  2304      // C*9
#define KSTEPS 72      // CK / 32
#define NT   64        // pixel tile per workgroup
#define GNCNT (16.0f * 9216.0f)   // elements per (batch, group)

static __device__ __forceinline__ unsigned short bf16_rne(float f) {
    unsigned int u = __float_as_uint(f);
    unsigned int r = (u + 0x7FFFu + ((u >> 16) & 1u)) >> 16;
    return (unsigned short)r;
}
static __device__ __forceinline__ float bf16_to_f32(unsigned short h) {
    return __uint_as_float(((unsigned int)h) << 16);
}

// ---------------------------------------------------------------------------
// Kernel 0: split weights into bf16 hi/lo in WMMA A-fragment layout; zero stats.
// A 16x32 bf16 fragment (ISA 7.12.2): lanes 0-15: M=lane, VGPR0-3 hold K=0..7
// (pairs), VGPR4-7 K=16..23; lanes 16-31: M=lane-16, K=8..15 and K=24..31.
// ---------------------------------------------------------------------------
__global__ __launch_bounds__(256) void prep_weights(
    const float* __restrict__ w, unsigned short* __restrict__ whi,
    unsigned short* __restrict__ wlo, float* __restrict__ stats)
{
    int t = blockIdx.x * 256 + threadIdx.x;
    if (blockIdx.x == 0 && threadIdx.x < 64) stats[threadIdx.x] = 0.0f;
    if (t >= Oc * CKc) return;
    int o  = t / CKc;
    int ck = t - o * CKc;
    float val = w[t];
    unsigned short h = bf16_rne(val);
    unsigned short l = bf16_rne(val - bf16_to_f32(h));
    int ot = o >> 4, m = o & 15;
    int s  = ck >> 5, r2 = ck & 31;
    int blk = r2 >> 4, q = r2 & 15;
    int lane, v;
    if (q < 8) { lane = m;      v = blk * 4 + (q >> 1); }
    else       { lane = 16 + m; v = blk * 4 + ((q - 8) >> 1); }
    int sub = q & 1;
    int dst = (((ot * KSTEPS + s) * 32 + lane) << 4) + v * 2 + sub;
    whi[dst] = h;
    wlo[dst] = l;
}

// ---------------------------------------------------------------------------
// Kernel 1: fused deformable sampling + WMMA GEMM + bias + GN partial stats.
// Grid: 2 batches * 144 pixel tiles of 64. Block: 256 threads = 8 waves.
// Wave w owns o-tiles {2w, 2w+1} x 4 pixel groups -> 8 fp32 16x16 accumulators.
// ---------------------------------------------------------------------------
__global__ __launch_bounds__(256) void mdcn_wmma(
    const float* __restrict__ x, const float* __restrict__ offset,
    const float* __restrict__ mask, const unsigned short* __restrict__ whi,
    const unsigned short* __restrict__ wlo, const float* __restrict__ bias,
    float* __restrict__ convout, float* __restrict__ stats)
{
    __shared__ int4   smp_idx[9 * NT];                 // 9216 B
    __shared__ float4 smp_w[9 * NT];                   // 9216 B
    __shared__ unsigned short VBhi[2][4 * 32 * 16];    // 2 x 4096 B (double buf)
    __shared__ unsigned short VBlo[2][4 * 32 * 16];    // 2 x 4096 B

    const int tile = blockIdx.x;
    const int b    = tile / 144;
    const int p0   = (tile - b * 144) * NT;
    const int tid  = threadIdx.x;
    const int lane = tid & 31;
    const int wid  = tid >> 5;

    // ---- Phase 1: per-(k,pixel) bilinear indices & mask-folded weights ----
    {
        const float* offs  = offset + b * 18 * HWc;
        const float* maskp = mask   + b * 9  * HWc;
        for (int j = tid; j < 9 * NT; j += 256) {
            int kk = j >> 6, p = j & (NT - 1);
            int pix = p0 + p;
            int hh = pix / Wc, ww = pix - hh * Wc;
            float oy = offs[(kk * 2 + 0) * HWc + pix];
            float ox = offs[(kk * 2 + 1) * HWc + pix];
            float mm = maskp[kk * HWc + pix];
            float py = (float)(hh + (kk / 3) - 1) + oy;
            float px = (float)(ww + (kk % 3) - 1) + ox;
            float fy = floorf(py), fx = floorf(px);
            float dy = py - fy,    dx = px - fx;
            int y0 = (int)fy, x0 = (int)fx;
            float w00 = (1.f - dy) * (1.f - dx) * mm;
            float w01 = (1.f - dy) * dx * mm;
            float w10 = dy * (1.f - dx) * mm;
            float w11 = dy * dx * mm;
            int4 id; float4 wt;
            {
                int yi = y0, xi = x0;
                bool v = (yi >= 0) & (yi < Hc) & (xi >= 0) & (xi < Wc);
                int yc = yi < 0 ? 0 : (yi > Hc - 1 ? Hc - 1 : yi);
                int xc = xi < 0 ? 0 : (xi > Wc - 1 ? Wc - 1 : xi);
                id.x = yc * Wc + xc; wt.x = v ? w00 : 0.f;
            }
            {
                int yi = y0, xi = x0 + 1;
                bool v = (yi >= 0) & (yi < Hc) & (xi >= 0) & (xi < Wc);
                int yc = yi < 0 ? 0 : (yi > Hc - 1 ? Hc - 1 : yi);
                int xc = xi < 0 ? 0 : (xi > Wc - 1 ? Wc - 1 : xi);
                id.y = yc * Wc + xc; wt.y = v ? w01 : 0.f;
            }
            {
                int yi = y0 + 1, xi = x0;
                bool v = (yi >= 0) & (yi < Hc) & (xi >= 0) & (xi < Wc);
                int yc = yi < 0 ? 0 : (yi > Hc - 1 ? Hc - 1 : yi);
                int xc = xi < 0 ? 0 : (xi > Wc - 1 ? Wc - 1 : xi);
                id.z = yc * Wc + xc; wt.z = v ? w10 : 0.f;
            }
            {
                int yi = y0 + 1, xi = x0 + 1;
                bool v = (yi >= 0) & (yi < Hc) & (xi >= 0) & (xi < Wc);
                int yc = yi < 0 ? 0 : (yi > Hc - 1 ? Hc - 1 : yi);
                int xc = xi < 0 ? 0 : (xi > Wc - 1 ? Wc - 1 : xi);
                id.w = yc * Wc + xc; wt.w = v ? w11 : 0.f;
            }
            smp_idx[j] = id;
            smp_w[j]   = wt;
        }
    }

    const int ot0   = wid * 2;
    const int pfix  = tid & (NT - 1);   // fixed pixel per thread in fill phase
    const int rb    = tid >> 6;         // 0..3
    const int pgfix = pfix >> 4;
    const int nfix  = pfix & 15;
    const float* xbase = x + (b << 8) * HWc;

    // issue the 32 global gathers for chunk s into registers (stay in flight)
    auto gather = [&](int s, float g[8][4]) {
#pragma unroll
        for (int i = 0; i < 8; i++) {
            int r  = rb + i * 4;
            int ck = s * 32 + r;
            int c  = ck / 9;
            int kk = ck - c * 9;
            const float* xc = xbase + c * HWc;
            int4 id = smp_idx[kk * NT + pfix];
            g[i][0] = xc[id.x];
            g[i][1] = xc[id.y];
            g[i][2] = xc[id.z];
            g[i][3] = xc[id.w];
        }
    };
    // combine with bilinear weights, split into bf16 hi/lo, store to LDS in
    // B-fragment layout: lanes 0-15 hold K=0..7,16..23, lanes 16-31 hold K=8..15,24..31.
    auto convert_store = [&](int s, const float g[8][4],
                             unsigned short* bhq, unsigned short* blq) {
#pragma unroll
        for (int i = 0; i < 8; i++) {
            int r  = rb + i * 4;
            int ck = s * 32 + r;
            int c  = ck / 9;
            int kk = ck - c * 9;
            float4 wt = smp_w[kk * NT + pfix];
            float val = wt.x * g[i][0] + wt.y * g[i][1] +
                        wt.z * g[i][2] + wt.w * g[i][3];
            unsigned short h = bf16_rne(val);
            unsigned short l = bf16_rne(val - bf16_to_f32(h));
            int rhalf = (r >> 3) & 1, rr = (r & 7) | ((r >> 4) << 3);
            int dst = (((pgfix << 5) + (rhalf << 4) + nfix) << 4) + rr;
            bhq[dst] = h;
            blq[dst] = l;
        }
    };

    v8f acc[2][4];
#pragma unroll
    for (int i = 0; i < 2; i++)
#pragma unroll
        for (int pg = 0; pg < 4; pg++)
#pragma unroll
            for (int e = 0; e < 8; e++) acc[i][pg][e] = 0.0f;

    // ---- prologue: fill chunk 0 ----
    __syncthreads();                    // smp tables visible to all waves
    float g[8][4];
    gather(0, g);
    convert_store(0, g, VBhi[0], VBlo[0]);
    __syncthreads();

    // ---- pipelined K-loop: gathers for s+1 overlap WMMAs of s ----
    for (int s = 0; s < KSTEPS; s++) {
        const int cur = s & 1;
        if (s + 1 < KSTEPS) gather(s + 1, g);      // loads in flight

        // A fragments for this wave's two o-tiles (32 B/lane, from L2)
        v16bf ah[2], al[2];
#pragma unroll
        for (int i = 0; i < 2; i++) {
            int aoff = (((ot0 + i) * KSTEPS + s) * 32 + lane) << 4;
            ah[i] = *reinterpret_cast<const v16bf*>(&whi[aoff]);
            al[i] = *reinterpret_cast<const v16bf*>(&wlo[aoff]);
        }
        if (s + 1 < KSTEPS) {   // prefetch next weight chunk -> global_prefetch_b8
            int pf = ((ot0 * KSTEPS + s + 1) * 32 + lane) << 4;
            __builtin_prefetch(&whi[pf], 0, 3);
            __builtin_prefetch(&wlo[pf], 0, 3);
        }

        // bf16x3 WMMAs on the current buffer
#pragma unroll
        for (int pg = 0; pg < 4; pg++) {
            v16bf bh = *reinterpret_cast<const v16bf*>(
                &VBhi[cur][((pg << 5) + lane) << 4]);
            v16bf bl = *reinterpret_cast<const v16bf*>(
                &VBlo[cur][((pg << 5) + lane) << 4]);
#pragma unroll
            for (int i = 0; i < 2; i++) {
                acc[i][pg] = __builtin_amdgcn_wmma_f32_16x16x32_bf16(
                    false, ah[i], false, bh, (short)0, acc[i][pg], false, false);
                acc[i][pg] = __builtin_amdgcn_wmma_f32_16x16x32_bf16(
                    false, al[i], false, bh, (short)0, acc[i][pg], false, false);
                acc[i][pg] = __builtin_amdgcn_wmma_f32_16x16x32_bf16(
                    false, ah[i], false, bl, (short)0, acc[i][pg], false, false);
            }
        }

        if (s + 1 < KSTEPS)
            convert_store(s + 1, g, VBhi[cur ^ 1], VBlo[cur ^ 1]);
        __syncthreads();   // next buffer ready; current buffer free to rewrite
    }

    // ---- Phase 3: bias, store conv output, GN partial sums (group == o-tile) ----
    // C-frag mapping: VGPR r, lane L -> M = r + 8*(L/16), N = L%16.
    const int mofs = (lane >> 4) * 8;
    const int ncol = lane & 15;
#pragma unroll
    for (int i = 0; i < 2; i++) {
        int ot = ot0 + i;
        float s1 = 0.f, s2 = 0.f;
#pragma unroll
        for (int pg = 0; pg < 4; pg++) {
            int pcol = p0 + pg * 16 + ncol;
#pragma unroll
            for (int r = 0; r < 8; r++) {
                int o = ot * 16 + r + mofs;
                float v = acc[i][pg][r] + bias[o];
                convout[((b << 8) + o) * HWc + pcol] = v;
                s1 += v;
                s2 += v * v;
            }
        }
        // wave32 butterfly reduction
#pragma unroll
        for (int d = 16; d >= 1; d >>= 1) {
            s1 += __shfl_xor(s1, d, 32);
            s2 += __shfl_xor(s2, d, 32);
        }
        if (lane == 0) {
            atomicAdd(&stats[((b << 4) + ot) * 2 + 0], s1);
            atomicAdd(&stats[((b << 4) + ot) * 2 + 1], s2);
        }
    }
}

// ---------------------------------------------------------------------------
// Kernel 2: GroupNorm apply (float4 elementwise), 38 MB streamed.
// ---------------------------------------------------------------------------
__global__ __launch_bounds__(256) void gn_apply(
    const float* __restrict__ conv, const float* __restrict__ stats,
    const float* __restrict__ gamma, const float* __restrict__ beta,
    float* __restrict__ out)
{
    int t = blockIdx.x * 256 + threadIdx.x;
    int e = t * 4;
    if (e >= 2 * Oc * HWc) return;
    int bo = e / HWc;
    int b = bo >> 8, o = bo & 255, g = o >> 4;
    float s1 = stats[((b << 4) + g) * 2 + 0];
    float s2 = stats[((b << 4) + g) * 2 + 1];
    float mean = s1 / GNCNT;
    float var  = s2 / GNCNT - mean * mean;
    float rs   = rsqrtf(var + 1e-5f);
    float ga = gamma[o] * rs;
    float be = beta[o] - mean * ga;
    float4 v = *reinterpret_cast<const float4*>(conv + e);
    float4 r;
    r.x = v.x * ga + be;
    r.y = v.y * ga + be;
    r.z = v.z * ga + be;
    r.w = v.w * ga + be;
    *reinterpret_cast<float4*>(out + e) = r;
}

// ---------------------------------------------------------------------------
extern "C" void kernel_launch(void* const* d_in, const int* in_sizes, int n_in,
                              void* d_out, int out_size, void* d_ws, size_t ws_size,
                              hipStream_t stream) {
    const float* x      = (const float*)d_in[0];
    const float* offset = (const float*)d_in[1];
    const float* mask   = (const float*)d_in[2];
    const float* weight = (const float*)d_in[3];
    const float* bias   = (const float*)d_in[4];
    const float* gamma  = (const float*)d_in[5];
    const float* beta   = (const float*)d_in[6];
    float* out = (float*)d_out;

    char* ws = (char*)d_ws;
    const size_t convBytes = (size_t)2 * Oc * HWc * sizeof(float);      // 18,874,368
    const size_t wBytes    = (size_t)Oc * CKc * sizeof(unsigned short); // 1,179,648
    float*          convout = (float*)ws;
    unsigned short* whi     = (unsigned short*)(ws + convBytes);
    unsigned short* wlo     = (unsigned short*)(ws + convBytes + wBytes);
    float*          stats   = (float*)(ws + convBytes + 2 * wBytes);

    prep_weights<<<(Oc * CKc) / 256, 256, 0, stream>>>(weight, whi, wlo, stats);
    mdcn_wmma<<<2 * (HWc / NT), 256, 0, stream>>>(
        x, offset, mask, whi, wlo, bias, convout, stats);
    gn_apply<<<(2 * Oc * HWc) / (256 * 4), 256, 0, stream>>>(
        convout, stats, gamma, beta, out);
}
